// ColbertNegativeCELoss_27822798143932
// MI455X (gfx1250) — compile-verified, matrix operand order checked by reading, not run on hardware
//
#include <hip/hip_runtime.h>
#include <hip/hip_bf16.h>

// ---------------------------------------------------------------------------
// ColBERT negative-CE loss for MI455X (gfx1250).
// Dominant work: 128x128 all-pairs MaxSim (34.4 GFLOP) -> matrix-core bound.
// Whole problem (34 MB incl. split-bf16 copies) lives in the 192 MB L2, so
// fragments are read straight from global; LDS only holds a 512 B reduction
// buffer. Precision: split-bf16 (hi+lo) with 3x v_wmma_f32_16x16x32_bf16 per
// K=32 step -> ~fp32 accuracy at ~5x the fp32-WMMA (16x16x4) instruction rate.
// ---------------------------------------------------------------------------

#define B_   128
#define NQ_  32
#define ND_  256
#define D_   128
#define INV_T 50.0f   // 1 / 0.02

typedef __attribute__((ext_vector_type(16))) __bf16 v16bf;
typedef __attribute__((ext_vector_type(8)))  float  v8f;

union FragU { uint4 q[2]; v16bf v; };

// Load one 16-lane-layout bf16 fragment: per lane, 8 contiguous bf16 at p
// (K = k0..k0+7) and 8 more at p+16 (K = k0+16..k0+23). Matches the CDNA5
// 16-bit 16x32 A layout (and B^T loaded from row-major docs).
__device__ __forceinline__ v16bf load_frag(const unsigned short* p) {
    FragU f;
    f.q[0] = *(const uint4*)(p);
    f.q[1] = *(const uint4*)(p + 16);
    return f.v;
}

__device__ __forceinline__ v8f wmma_bf16(v16bf a, v16bf b, v8f c) {
    return __builtin_amdgcn_wmma_f32_16x16x32_bf16(
        /*neg_a=*/false, a, /*neg_b=*/false, b,
        /*c_mod=*/(short)0, c, /*reuse_a=*/false, /*reuse_b=*/false);
}

// Computes sum_{n<32} max_{s<256} dot(q[n,:], d[s,:]) for one (query, doc)
// pair using one 256-thread block (8 waves). Result valid on threadIdx.x==0.
// rowred: 128 floats of LDS scratch.
__device__ float pair_maxsim_score(const unsigned short* __restrict__ qhi,
                                   const unsigned short* __restrict__ qlo,
                                   const unsigned short* __restrict__ dhi,
                                   const unsigned short* __restrict__ dlo,
                                   float* rowred) {
    const int tid  = threadIdx.x;
    const int wave = tid >> 5;
    const int lane = tid & 31;
    const int half = lane >> 4;   // K sub-block select per ISA layout
    const int l    = lane & 15;
    const int mt   = wave & 1;    // m-tile: rows mt*16 .. mt*16+15
    const int ng   = wave >> 1;   // n-group: 4 n-tiles = 64 doc tokens

    // Hoist A (query) fragments: 4 k-steps, hi+lo.
    v16bf ah[4], al[4];
    const int arow = mt * 16 + l;
    #pragma unroll
    for (int ks = 0; ks < 4; ++ks) {
        const int k0 = ks * 32 + half * 8;
        ah[ks] = load_frag(qhi + arow * D_ + k0);
        al[ks] = load_frag(qlo + arow * D_ + k0);
    }

    float rm[8];
    #pragma unroll
    for (int i = 0; i < 8; ++i) rm[i] = -1e30f;

    #pragma unroll
    for (int nt = 0; nt < 4; ++nt) {
        const int nrow = (ng * 4 + nt) * 16 + l;     // doc token index
        const unsigned short* dh = dhi + nrow * D_;
        const unsigned short* dl = dlo + nrow * D_;
        v8f acc = {};
        #pragma unroll
        for (int ks = 0; ks < 4; ++ks) {
            const int k0 = ks * 32 + half * 8;
            const v16bf bh = load_frag(dh + k0);
            const v16bf bl = load_frag(dl + k0);
            // small cross terms first, then the dominant hi*hi
            acc = wmma_bf16(al[ks], bh, acc);
            acc = wmma_bf16(ah[ks], bl, acc);
            acc = wmma_bf16(ah[ks], bh, acc);
        }
        #pragma unroll
        for (int i = 0; i < 8; ++i) rm[i] = fmaxf(rm[i], acc[i]);
    }

    // C layout: VGPR r -> row (r + 8*half), lane%16 -> column. Max over the
    // 16 columns = shuffle-max within each 16-lane half.
    #pragma unroll
    for (int off = 1; off < 16; off <<= 1) {
        #pragma unroll
        for (int i = 0; i < 8; ++i)
            rm[i] = fmaxf(rm[i], __shfl_xor(rm[i], off, 32));
    }
    if (lane == 0) {
        #pragma unroll
        for (int i = 0; i < 8; ++i) rowred[wave * 16 + i] = rm[i];
    } else if (lane == 16) {
        #pragma unroll
        for (int i = 0; i < 8; ++i) rowred[wave * 16 + 8 + i] = rm[i];
    }
    __syncthreads();

    float result = 0.0f;
    if (tid < 32) {                      // wave 0: finish max over n-groups,
        const int row = tid;             // then sum over the 32 query tokens
        const int rmt = row >> 4, r16 = row & 15;
        float v = -1e30f;
        #pragma unroll
        for (int g = 0; g < 4; ++g)
            v = fmaxf(v, rowred[(g * 2 + rmt) * 16 + r16]);
        #pragma unroll
        for (int off = 16; off >= 1; off >>= 1)
            v += __shfl_xor(v, off, 32);
        result = v;
    }
    __syncthreads();                     // rowred reusable by caller's next pass
    return result;
}

// ---- Kernel 0: split fp32 -> (bf16 hi, bf16 lo), both RNE-rounded ----------
__global__ void __launch_bounds__(256)
split_bf16_kernel(const float* __restrict__ src, unsigned short* __restrict__ hi,
                  unsigned short* __restrict__ lo, int n) {
    const int i = blockIdx.x * blockDim.x + threadIdx.x;
    if (i >= n) return;
    const float x = src[i];
    unsigned int u = __float_as_uint(x);
    unsigned int r = u + 0x7FFFu + ((u >> 16) & 1u);          // RNE to bf16
    const unsigned short h = (unsigned short)(r >> 16);
    const float hf = __uint_as_float((unsigned int)h << 16);
    const float res = x - hf;
    unsigned int ul = __float_as_uint(res);
    unsigned int rl = ul + 0x7FFFu + ((ul >> 16) & 1u);
    hi[i] = h;
    lo[i] = (unsigned short)(rl >> 16);
}

// ---- Kernel 1: all-pairs MaxSim scores S[b][c] -----------------------------
__global__ void __launch_bounds__(256)
allpairs_kernel(const unsigned short* __restrict__ qhi,
                const unsigned short* __restrict__ qlo,
                const unsigned short* __restrict__ dhi,
                const unsigned short* __restrict__ dlo,
                float* __restrict__ S) {
    __shared__ float rowred[128];
    const int c = blockIdx.x;
    const int b = blockIdx.y;
    const float s = pair_maxsim_score(qhi + (size_t)b * NQ_ * D_,
                                      qlo + (size_t)b * NQ_ * D_,
                                      dhi + (size_t)c * ND_ * D_,
                                      dlo + (size_t)c * ND_ * D_, rowred);
    if (threadIdx.x == 0) S[b * B_ + c] = s;
}

// ---- Kernel 2: diagonal MaxSim vs negative docs ----------------------------
__global__ void __launch_bounds__(256)
neg_kernel(const unsigned short* __restrict__ qhi,
           const unsigned short* __restrict__ qlo,
           const unsigned short* __restrict__ nhi,
           const unsigned short* __restrict__ nlo,
           float* __restrict__ neg) {
    __shared__ float rowred[128];
    const int b = blockIdx.x;
    const float s = pair_maxsim_score(qhi + (size_t)b * NQ_ * D_,
                                      qlo + (size_t)b * NQ_ * D_,
                                      nhi + (size_t)b * ND_ * D_,
                                      nlo + (size_t)b * ND_ * D_, rowred);
    if (threadIdx.x == 0) neg[b] = s;
}

// ---- Kernel 3: softplus + CE finalization ----------------------------------
__global__ void __launch_bounds__(128)
finalize_kernel(const float* __restrict__ S, const float* __restrict__ neg,
                float* __restrict__ out) {
    __shared__ float red[B_];
    const int b = threadIdx.x;
    const float pos = S[b * B_ + b];

    // softplus((neg - pos) / t), overflow-safe
    const float x = (neg[b] - pos) * INV_T;
    float sp;
    if (x > 20.0f)       sp = x;
    else if (x < -20.0f) sp = expf(x);
    else                 sp = log1pf(expf(x));

    // cross-entropy row term: logsumexp_c(S[b,c]/t) - S[b,b]/t
    float ms = -1e30f;
    for (int c = 0; c < B_; ++c) ms = fmaxf(ms, S[b * B_ + c]);
    const float mz = ms * INV_T;
    float sum = 0.0f;
    for (int c = 0; c < B_; ++c) sum += expf(S[b * B_ + c] * INV_T - mz);
    const float ce = (mz + logf(sum)) - pos * INV_T;

    red[b] = sp + ce;
    __syncthreads();
    if (b == 0) {
        float t = 0.0f;
        for (int i = 0; i < B_; ++i) t += red[i];
        out[0] = t * (0.5f / (float)B_);
    }
}

extern "C" void kernel_launch(void* const* d_in, const int* in_sizes, int n_in,
                              void* d_out, int out_size, void* d_ws, size_t ws_size,
                              hipStream_t stream) {
    const float* q  = (const float*)d_in[0];   // [128,32,128]
    const float* dd = (const float*)d_in[1];   // [128,256,128]
    const float* nd = (const float*)d_in[2];   // [128,256,128]

    const int QN = B_ * NQ_ * D_;   // 524288
    const int DN = B_ * ND_ * D_;   // 4194304

    // Workspace layout (~34.1 MB): split-bf16 copies + scores.
    char* p = (char*)d_ws;
    unsigned short* qhi = (unsigned short*)p; p += (size_t)QN * 2;
    unsigned short* qlo = (unsigned short*)p; p += (size_t)QN * 2;
    unsigned short* dhi = (unsigned short*)p; p += (size_t)DN * 2;
    unsigned short* dlo = (unsigned short*)p; p += (size_t)DN * 2;
    unsigned short* nhi = (unsigned short*)p; p += (size_t)DN * 2;
    unsigned short* nlo = (unsigned short*)p; p += (size_t)DN * 2;
    float* S   = (float*)p; p += (size_t)B_ * B_ * 4;
    float* neg = (float*)p; p += (size_t)B_ * 4;

    split_bf16_kernel<<<(QN + 255) / 256, 256, 0, stream>>>(q,  qhi, qlo, QN);
    split_bf16_kernel<<<(DN + 255) / 256, 256, 0, stream>>>(dd, dhi, dlo, DN);
    split_bf16_kernel<<<(DN + 255) / 256, 256, 0, stream>>>(nd, nhi, nlo, DN);

    allpairs_kernel<<<dim3(B_, B_), 256, 0, stream>>>(qhi, qlo, dhi, dlo, S);
    neg_kernel<<<B_, 256, 0, stream>>>(qhi, qlo, nhi, nlo, neg);
    finalize_kernel<<<1, B_, 0, stream>>>(S, neg, (float*)d_out);
}
